// EncoderLayer_19361712571367
// MI455X (gfx1250) — matrix operand order repaired and checked by reading, not compiled, not run on hardware
//
#include <hip/hip_runtime.h>
#include <hip/hip_bf16.h>

typedef __attribute__((ext_vector_type(16))) _Float16 v16h;
typedef __attribute__((ext_vector_type(8)))  float    v8f;

#define D_MODEL 1024
#define N_HEADS 16
#define D_HEAD  64
#define D_FF    4096
#define BATCH   4
#define SEQ     2048
#define ROWS    (BATCH * SEQ)   // 8192

union F16Frag { uint4 u[2]; v16h v; };
union F16x8   { uint4 u; _Float16 e[8]; };
union F32x4   { uint4 u; float e[4]; };

__device__ inline v16h load_frag2(const _Float16* p0, const _Float16* p1) {
  F16Frag t;
  t.u[0] = *reinterpret_cast<const uint4*>(p0);
  t.u[1] = *reinterpret_cast<const uint4*>(p1);
  return t.v;
}

__device__ inline v8f wmma_f16(v16h a, v16h b, v8f c) {
  return __builtin_amdgcn_wmma_f32_16x16x32_f16(
      /*neg_a=*/false, a, /*neg_b=*/false, b,
      /*c_mod=*/(short)0, c, /*reuse_a=*/false, /*reuse_b=*/false);
}

// ---------------------------------------------------------------- pack kernels

__global__ void cast_f32_f16(const float* __restrict__ src,
                             _Float16* __restrict__ dst, int n) {
  for (int i = blockIdx.x * blockDim.x + threadIdx.x; i < n;
       i += gridDim.x * blockDim.x)
    dst[i] = (_Float16)src[i];
}

// src f32 [R][C] -> dst f16 [C][R] (transpose-cast, one-time)
__global__ void cast_transpose_f16(const float* __restrict__ src,
                                   _Float16* __restrict__ dst, int R, int C) {
  for (int i = blockIdx.x * blockDim.x + threadIdx.x; i < R * C;
       i += gridDim.x * blockDim.x) {
    int r = i / C, c = i % C;
    dst[(size_t)c * R + r] = (_Float16)src[i];
  }
}

// Wq/Wk/Wv [H, D_MODEL, D_HEAD] -> W^T f16 [3072][1024], row n = p*1024+h*64+d
__global__ void pack_qkv_wT(const float* __restrict__ Wq,
                            const float* __restrict__ Wk,
                            const float* __restrict__ Wv,
                            _Float16* __restrict__ out) {
  int id = blockIdx.x * blockDim.x + threadIdx.x;
  const int per = N_HEADS * D_MODEL * D_HEAD;   // 1048576
  if (id >= 3 * per) return;
  int p   = id / per;
  int rem = id % per;
  int h = rem / (D_MODEL * D_HEAD);
  int m = (rem / D_HEAD) % D_MODEL;
  int d = rem % D_HEAD;
  const float* W = (p == 0) ? Wq : ((p == 1) ? Wk : Wv);
  out[(size_t)(p * 1024 + h * 64 + d) * 1024 + m] =
      (_Float16)W[((size_t)h * D_MODEL + m) * D_HEAD + d];
}

__global__ void pack_qkv_b(const float* __restrict__ bq,
                           const float* __restrict__ bk,
                           const float* __restrict__ bv,
                           float* __restrict__ out) {
  int i = blockIdx.x * blockDim.x + threadIdx.x;
  if (i >= 3072) return;
  int p = i >> 10, r = i & 1023;
  out[i] = (p == 0 ? bq : (p == 1 ? bk : bv))[r];
}

// ---------------------------------------------------------------- WMMA GEMM
// C^T = W^T(NxK) * act^T(KxM), i.e. C[m][n] = sum_k act[m][k]*W[k][n].
// NO LDS: both operands stream as contiguous 16B fragment chunks from
// global/L2 (whole working set is L2-resident on 192MB L2). Block = 8 waves,
// tile 256n x 128m; wave tile 64n x 64m (16 WMMAs per 8 fragment loads).
// Epilogue: lane holds 8 consecutive n -> fully vectorized b128 stores.

template <bool RELU, bool OUTF32, bool OUTF16>
__global__ void __launch_bounds__(256)
gemm_ct(const _Float16* __restrict__ WT, const _Float16* __restrict__ act,
        const float* __restrict__ bias, float* __restrict__ Cf,
        _Float16* __restrict__ Ch, int M, int N, int K) {
  const int tid = threadIdx.x, wave = tid >> 5, lane = tid & 31;
  const int wn = wave >> 1, wm = wave & 1, hi = lane >> 4, ln = lane & 15;
  const int n0 = blockIdx.y * 256 + wn * 64;
  const int m0 = blockIdx.x * 128 + wm * 64;

  const v8f vz = {0.f, 0.f, 0.f, 0.f, 0.f, 0.f, 0.f, 0.f};
  v8f acc[4][4];
#pragma unroll
  for (int i = 0; i < 4; i++)
#pragma unroll
    for (int j = 0; j < 4; j++) acc[i][j] = vz;

  const _Float16* wrow[4];
  const _Float16* arow[4];
#pragma unroll
  for (int t = 0; t < 4; t++) {
    wrow[t] = WT + (size_t)(n0 + t * 16 + ln) * K;
    arow[t] = act + (size_t)(m0 + t * 16 + ln) * K;
  }

  for (int k0 = 0; k0 < K; k0 += 32) {
    if (k0 + 64 < K) {  // WGP-scope prefetch of upcoming K-chunks
      __builtin_prefetch(wrow[0] + k0 + 64, 0, 3);
      __builtin_prefetch(wrow[2] + k0 + 64, 0, 3);
      __builtin_prefetch(arow[0] + k0 + 64, 0, 3);
      __builtin_prefetch(arow[2] + k0 + 64, 0, 3);
    }
    // A frag (W^T rows): row n; halves j<8 -> k=hi*8+j, j>=8 -> k=16+hi*8+(j-8)
    v16h af[4];
#pragma unroll
    for (int t = 0; t < 4; t++)
      af[t] = load_frag2(wrow[t] + k0 + hi * 8, wrow[t] + k0 + 16 + hi * 8);
    // B frag (act rows): col m; halves j<8 -> k=hi*8+j, j>=8 -> k=16+hi*8+(j-8)
    v16h bf[4];
#pragma unroll
    for (int t = 0; t < 4; t++)
      bf[t] = load_frag2(arow[t] + k0 + hi * 8, arow[t] + k0 + 16 + hi * 8);
#pragma unroll
    for (int t4 = 0; t4 < 4; t4++)
#pragma unroll
      for (int t2 = 0; t2 < 4; t2++)
        acc[t4][t2] = wmma_f16(af[t4], bf[t2], acc[t4][t2]);
  }

  // epilogue: D rows = n (hi*8+r), cols = m (ln): 8 consecutive n per lane
#pragma unroll
  for (int t4 = 0; t4 < 4; t4++) {
    int nbase = n0 + t4 * 16 + hi * 8;
    F32x4 b0, b1;
    b0.u = *reinterpret_cast<const uint4*>(bias + nbase);
    b1.u = *reinterpret_cast<const uint4*>(bias + nbase + 4);
#pragma unroll
    for (int t2 = 0; t2 < 4; t2++) {
      int m = m0 + t2 * 16 + ln;
      float y[8];
#pragma unroll
      for (int r = 0; r < 8; r++) {
        y[r] = acc[t4][t2][r] + (r < 4 ? b0.e[r] : b1.e[r - 4]);
        if (RELU) y[r] = fmaxf(y[r], 0.f);
      }
      if (OUTF32) {
        F32x4 o0, o1;
#pragma unroll
        for (int r = 0; r < 4; r++) { o0.e[r] = y[r]; o1.e[r] = y[r + 4]; }
        uint4* dst = reinterpret_cast<uint4*>(Cf + (size_t)m * N + nbase);
        dst[0] = o0.u; dst[1] = o1.u;
      }
      if (OUTF16) {
        F16x8 oh;
#pragma unroll
        for (int r = 0; r < 8; r++) oh.e[r] = (_Float16)y[r];
        *reinterpret_cast<uint4*>(Ch + (size_t)m * N + nbase) = oh.u;
      }
    }
  }
}

// QKV projection, C^T form, N=3072, K=1024. Epilogue: Q|K columns (n<2048)
// packed b128 into qk[m][2048]; V columns written pre-transposed into
// vT[b*1024 + (n-2048)][s] so attention can load V fragments directly.
__global__ void __launch_bounds__(256)
gemm_qkv(const _Float16* __restrict__ WT, const _Float16* __restrict__ act,
         const float* __restrict__ bias, _Float16* __restrict__ qk,
         _Float16* __restrict__ vT) {
  const int K = 1024;
  const int tid = threadIdx.x, wave = tid >> 5, lane = tid & 31;
  const int wn = wave >> 1, wm = wave & 1, hi = lane >> 4, ln = lane & 15;
  const int n0 = blockIdx.y * 256 + wn * 64;
  const int m0 = blockIdx.x * 128 + wm * 64;

  const v8f vz = {0.f, 0.f, 0.f, 0.f, 0.f, 0.f, 0.f, 0.f};
  v8f acc[4][4];
#pragma unroll
  for (int i = 0; i < 4; i++)
#pragma unroll
    for (int j = 0; j < 4; j++) acc[i][j] = vz;

  const _Float16* wrow[4];
  const _Float16* arow[4];
#pragma unroll
  for (int t = 0; t < 4; t++) {
    wrow[t] = WT + (size_t)(n0 + t * 16 + ln) * K;
    arow[t] = act + (size_t)(m0 + t * 16 + ln) * K;
  }

  for (int k0 = 0; k0 < K; k0 += 32) {
    if (k0 + 64 < K) {
      __builtin_prefetch(wrow[0] + k0 + 64, 0, 3);
      __builtin_prefetch(wrow[2] + k0 + 64, 0, 3);
      __builtin_prefetch(arow[0] + k0 + 64, 0, 3);
      __builtin_prefetch(arow[2] + k0 + 64, 0, 3);
    }
    v16h af[4], bf[4];
#pragma unroll
    for (int t = 0; t < 4; t++)
      af[t] = load_frag2(wrow[t] + k0 + hi * 8, wrow[t] + k0 + 16 + hi * 8);
#pragma unroll
    for (int t = 0; t < 4; t++)
      bf[t] = load_frag2(arow[t] + k0 + hi * 8, arow[t] + k0 + 16 + hi * 8);
#pragma unroll
    for (int t4 = 0; t4 < 4; t4++)
#pragma unroll
      for (int t2 = 0; t2 < 4; t2++)
        acc[t4][t2] = wmma_f16(af[t4], bf[t2], acc[t4][t2]);
  }

#pragma unroll
  for (int t4 = 0; t4 < 4; t4++) {
    int nbase = n0 + t4 * 16 + hi * 8;
    F32x4 b0, b1;
    b0.u = *reinterpret_cast<const uint4*>(bias + nbase);
    b1.u = *reinterpret_cast<const uint4*>(bias + nbase + 4);
#pragma unroll
    for (int t2 = 0; t2 < 4; t2++) {
      int m = m0 + t2 * 16 + ln;
      _Float16 yh[8];
#pragma unroll
      for (int r = 0; r < 8; r++)
        yh[r] = (_Float16)(acc[t4][t2][r] + (r < 4 ? b0.e[r] : b1.e[r - 4]));
      if (nbase < 2048) {           // Q|K: row-major packed store
        F16x8 oh;
#pragma unroll
        for (int r = 0; r < 8; r++) oh.e[r] = yh[r];
        *reinterpret_cast<uint4*>(qk + (size_t)m * 2048 + nbase) = oh.u;
      } else {                       // V: scatter into transposed vT[hd][s]
        int b = m >> 11, s = m & 2047;
#pragma unroll
        for (int r = 0; r < 8; r++)
          vT[((size_t)b * 1024 + (nbase + r - 2048)) * SEQ + s] = yh[r];
      }
    }
  }
}

// ---------------------------------------------------------------- flash attention
// qk: [B*S, 2048] f16 (Q|K); vT: [B][1024][S] f16 (pre-transposed V).
// Transposed scores (S^T = K * Q^T): per-query key scores live in one lane.
// Each wave processes TWO 16-query groups that share the same K and V
// fragments -> 16 WMMAs per 32-key tile for one set of K/V loads. No block
// barriers; only the wave-local P transpose goes through LDS.

__global__ void __launch_bounds__(128)
attn_flash(const _Float16* __restrict__ qk, const _Float16* __restrict__ vT,
           _Float16* __restrict__ o) {
  __shared__ __align__(16) _Float16 Ph[4][2][16 * 40];  // per wave, per q-group
  const int w = threadIdx.x >> 5, lane = threadIdx.x & 31;
  const int hi = lane >> 4, ln = lane & 15;
  const int b = blockIdx.y >> 4, h = blockIdx.y & 15;
  const int q0 = blockIdx.x * 128 + w * 32;
  const size_t RS = 2048;
  const _Float16* Qb = qk + ((size_t)b * SEQ + q0) * RS + h * 64;
  const _Float16* Kb = qk + ((size_t)b * SEQ) * RS + 1024 + h * 64;
  const _Float16* Vb = vT + ((size_t)b * 1024 + h * 64) * SEQ;

  // Q^T as B-fragments for both q-groups: lane n=q=ln, halves k=hi*8+j / 16+hi*8+(j-8)
  v16h bqf[2][2];
#pragma unroll
  for (int g = 0; g < 2; g++) {
    const _Float16* qr = Qb + (size_t)(g * 16 + ln) * RS;
    bqf[g][0] = load_frag2(qr + hi * 8, qr + 16 + hi * 8);
    bqf[g][1] = load_frag2(qr + 32 + hi * 8, qr + 48 + hi * 8);
  }

  const v8f vz = {0.f, 0.f, 0.f, 0.f, 0.f, 0.f, 0.f, 0.f};
  float m_cur[2] = {-1e30f, -1e30f}, l_cur[2] = {0.f, 0.f};
  v8f acc[2][4];
#pragma unroll
  for (int g = 0; g < 2; g++)
#pragma unroll
    for (int dt = 0; dt < 4; dt++) acc[g][dt] = vz;

  const float scale = 0.125f;  // 1/sqrt(64)
  v16h ap[2];

  for (int j = 0; j < SEQ; j += 32) {
    if (j + 32 < SEQ) {
      __builtin_prefetch(Kb + ((size_t)(j + 32 + ln)) * RS, 0, 3);
      __builtin_prefetch(Vb + (size_t)ln * SEQ + j + 32, 0, 3);
    }
    // K A-frags for this 32-key tile, shared across both q-groups
    v16h kf[4];
    {
      const _Float16* ka0 = Kb + ((size_t)(j + ln)) * RS;
      kf[0] = load_frag2(ka0 + hi * 8, ka0 + 16 + hi * 8);
      kf[1] = load_frag2(ka0 + 32 + hi * 8, ka0 + 48 + hi * 8);
      const _Float16* ka1 = Kb + ((size_t)(j + 16 + ln)) * RS;
      kf[2] = load_frag2(ka1 + hi * 8, ka1 + 16 + hi * 8);
      kf[3] = load_frag2(ka1 + 32 + hi * 8, ka1 + 48 + hi * 8);
    }

#pragma unroll
    for (int g = 0; g < 2; g++) {
      // S^T tiles: rows = keys (hi*8+r), cols = q (ln)
      v8f st0 = wmma_f16(kf[0], bqf[g][0], vz);
      st0 = wmma_f16(kf[1], bqf[g][1], st0);
      v8f st1 = wmma_f16(kf[2], bqf[g][0], vz);
      st1 = wmma_f16(kf[3], bqf[g][1], st1);

      // streaming softmax for column q=ln: in-lane reduce + one xor-16 shuffle
      float mraw = st0[0];
#pragma unroll
      for (int r = 1; r < 8; r++) mraw = fmaxf(mraw, st0[r]);
#pragma unroll
      for (int r = 0; r < 8; r++) mraw = fmaxf(mraw, st1[r]);
      float mxs = mraw * scale;
      mxs = fmaxf(mxs, __shfl_xor(mxs, 16, 32));
      float mn = fmaxf(m_cur[g], mxs);
      float corr = __expf(m_cur[g] - mn);
      float p0[8], p1[8], rs = 0.f;
#pragma unroll
      for (int r = 0; r < 8; r++) { p0[r] = __expf(st0[r] * scale - mn); rs += p0[r]; }
#pragma unroll
      for (int r = 0; r < 8; r++) { p1[r] = __expf(st1[r] * scale - mn); rs += p1[r]; }
      rs += __shfl_xor(rs, 16, 32);
      l_cur[g] = l_cur[g] * corr + rs;
      m_cur[g] = mn;

      // broadcast corr into O-accumulator row layout (row q = hi*8 + r)
      float corr_r[8];
#pragma unroll
      for (int r = 0; r < 8; r++) corr_r[r] = __shfl(corr, hi * 8 + r, 32);
#pragma unroll
      for (int dt = 0; dt < 4; dt++)
#pragma unroll
        for (int r = 0; r < 8; r++) acc[g][dt][r] = acc[g][dt][r] * corr_r[r];

      // pack P (already in A-fragment order): Ph[q][key], two b128 stores
      F16x8 pk0, pk1;
#pragma unroll
      for (int r = 0; r < 8; r++) { pk0.e[r] = (_Float16)p0[r]; pk1.e[r] = (_Float16)p1[r]; }
      _Float16* pw = &Ph[w][g][0];
      *reinterpret_cast<uint4*>(&pw[ln * 40 + hi * 8]) = pk0.u;
      *reinterpret_cast<uint4*>(&pw[ln * 40 + 16 + hi * 8]) = pk1.u;
      const _Float16* pr = pw + ln * 40;
      ap[g] = load_frag2(pr + hi * 8, pr + 16 + hi * 8);
    }

    // O += P(16q x 32k) * V(32k x 64d): V frags shared across both q-groups
#pragma unroll
    for (int dt = 0; dt < 4; dt++) {
      const _Float16* vp = Vb + (size_t)(dt * 16 + ln) * SEQ + j;
      v16h vf = load_frag2(vp + hi * 8, vp + 16 + hi * 8);
      acc[0][dt] = wmma_f16(ap[0], vf, acc[0][dt]);
      acc[1][dt] = wmma_f16(ap[1], vf, acc[1][dt]);
    }
  }

  // normalize: gather 1/l for row q = hi*8 + r, then write O (f16)
#pragma unroll
  for (int g = 0; g < 2; g++) {
    float invl = 1.f / l_cur[g];
    float inv_r[8];
#pragma unroll
    for (int r = 0; r < 8; r++) inv_r[r] = __shfl(invl, hi * 8 + r, 32);
#pragma unroll
    for (int dt = 0; dt < 4; dt++)
#pragma unroll
      for (int r = 0; r < 8; r++) {
        int row = q0 + g * 16 + hi * 8 + r;
        int d = dt * 16 + ln;
        o[((size_t)b * SEQ + row) * 1024 + h * 64 + d] =
            (_Float16)(acc[g][dt][r] * inv_r[r]);
      }
  }
}

// ---------------------------------------------------------------- layernorm

__global__ void __launch_bounds__(256)
layernorm_residual(const float* __restrict__ a, const float* __restrict__ resid,
                   const float* __restrict__ g, const float* __restrict__ be,
                   float* __restrict__ outf, _Float16* __restrict__ outh) {
  __shared__ float s1[256], s2[256];
  int row = blockIdx.x, t = threadIdx.x;
  const size_t base = (size_t)row * D_MODEL;
  float v[4], sum = 0.f, sq = 0.f;
#pragma unroll
  for (int k = 0; k < 4; k++) {
    int c = t + k * 256;
    float x = a[base + c] + resid[base + c];
    v[k] = x; sum += x; sq += x * x;
  }
  s1[t] = sum; s2[t] = sq;
  __syncthreads();
  for (int off = 128; off > 0; off >>= 1) {
    if (t < off) { s1[t] += s1[t + off]; s2[t] += s2[t + off]; }
    __syncthreads();
  }
  float mean = s1[0] * (1.0f / D_MODEL);
  float var = s2[0] * (1.0f / D_MODEL) - mean * mean;
  float rstd = rsqrtf(var + 1e-5f);
#pragma unroll
  for (int k = 0; k < 4; k++) {
    int c = t + k * 256;
    float y = (v[k] - mean) * rstd * g[c] + be[c];
    if (outf) outf[base + c] = y;
    if (outh) outh[base + c] = (_Float16)y;
  }
}

// ---------------------------------------------------------------- launch

extern "C" void kernel_launch(void* const* d_in, const int* in_sizes, int n_in,
                              void* d_out, int out_size, void* d_ws,
                              size_t ws_size, hipStream_t stream) {
  (void)in_sizes; (void)n_in; (void)out_size; (void)ws_size;
  const float* x   = (const float*)d_in[0];
  const float* Wq  = (const float*)d_in[1];
  const float* bq  = (const float*)d_in[2];
  const float* Wk  = (const float*)d_in[3];
  const float* bk  = (const float*)d_in[4];
  const float* Wv  = (const float*)d_in[5];
  const float* bv  = (const float*)d_in[6];
  const float* Wo  = (const float*)d_in[7];
  const float* bo  = (const float*)d_in[8];
  const float* W1  = (const float*)d_in[9];
  const float* b1  = (const float*)d_in[10];
  const float* W2  = (const float*)d_in[11];
  const float* b2  = (const float*)d_in[12];
  const float* g1  = (const float*)d_in[13];
  const float* be1 = (const float*)d_in[14];
  const float* g2  = (const float*)d_in[15];
  const float* be2 = (const float*)d_in[16];
  float* out = (float*)d_out;

  char* p = (char*)d_ws;
  auto alloc = [&](size_t bytes) {
    char* r = p;
    p += (bytes + 255) & ~(size_t)255;
    return r;
  };
  _Float16* xh    = (_Float16*)alloc((size_t)ROWS * 1024 * 2);
  _Float16* wqkvT = (_Float16*)alloc((size_t)3072 * 1024 * 2);
  float*    bqkv  = (float*)alloc((size_t)3072 * 4);
  _Float16* woT   = (_Float16*)alloc((size_t)1024 * 1024 * 2);
  _Float16* w1T   = (_Float16*)alloc((size_t)4096 * 1024 * 2);
  _Float16* w2T   = (_Float16*)alloc((size_t)1024 * 4096 * 2);
  _Float16* qkb   = (_Float16*)alloc((size_t)ROWS * 2048 * 2);
  _Float16* vT    = (_Float16*)alloc((size_t)BATCH * 1024 * SEQ * 2);
  _Float16* oat   = (_Float16*)alloc((size_t)ROWS * 1024 * 2);
  float*    mha   = (float*)alloc((size_t)ROWS * 1024 * 4);
  float*    n1f   = (float*)alloc((size_t)ROWS * 1024 * 4);
  _Float16* n1h   = (_Float16*)alloc((size_t)ROWS * 1024 * 2);
  _Float16* ff1   = (_Float16*)alloc((size_t)ROWS * 4096 * 2);
  float*    ff2   = (float*)alloc((size_t)ROWS * 1024 * 4);

  cast_f32_f16<<<4096, 256, 0, stream>>>(x, xh, ROWS * 1024);
  pack_qkv_wT<<<(3 * N_HEADS * D_MODEL * D_HEAD + 255) / 256, 256, 0, stream>>>(
      Wq, Wk, Wv, wqkvT);
  pack_qkv_b<<<12, 256, 0, stream>>>(bq, bk, bv, bqkv);
  cast_transpose_f16<<<2048, 256, 0, stream>>>(Wo, woT, 1024, 1024);
  cast_transpose_f16<<<4096, 256, 0, stream>>>(W1, w1T, 1024, 4096);
  cast_transpose_f16<<<4096, 256, 0, stream>>>(W2, w2T, 4096, 1024);

  // QKV projection with Q|K row-major and V transposed epilogue
  gemm_qkv<<<dim3(ROWS / 128, 3072 / 256), 256, 0, stream>>>(
      wqkvT, xh, bqkv, qkb, vT);

  // flash attention per (b,h), 128 query rows per block (32 per wave)
  attn_flash<<<dim3(SEQ / 128, BATCH * N_HEADS), 128, 0, stream>>>(qkb, vT, oat);

  // output projection -> f32
  gemm_ct<false, true, false><<<dim3(ROWS / 128, 1024 / 256), 256, 0, stream>>>(
      woT, oat, bo, mha, nullptr, ROWS, 1024, 1024);

  // norm1 = LN(mha + x)
  layernorm_residual<<<ROWS, 256, 0, stream>>>(mha, x, g1, be1, n1f, n1h);

  // FFN1: relu(norm1 @ W1 + b1) -> f16
  gemm_ct<true, false, true><<<dim3(ROWS / 128, 4096 / 256), 256, 0, stream>>>(
      w1T, n1h, b1, nullptr, ff1, ROWS, 4096, 1024);

  // FFN2: ff1 @ W2 + b2 -> f32
  gemm_ct<false, true, false><<<dim3(ROWS / 128, 1024 / 256), 256, 0, stream>>>(
      w2T, ff1, b2, ff2, nullptr, ROWS, 1024, 4096);

  // norm2 = LN(ff2 + norm1) -> d_out
  layernorm_residual<<<ROWS, 256, 0, stream>>>(ff2, n1f, g2, be2, out, nullptr);
}
